// DINOBevAligner_57123065037282
// MI455X (gfx1250) — matrix operand order repaired and checked by reading, not compiled, not run on hardware
//
#include <hip/hip_runtime.h>
#include <hip/hip_bf16.h>

// ---------------------------------------------------------------------------
// DINO BEV aligner for MI455X (gfx1250, wave32).
//  k1 build_mix : fold softmax(logits) grouped mix into dense 768x256 f16 mat
//  k2 ln_tokens : pre-LN of (6*1369) token rows, store f16 (halves gather B/W)
//  k3 fuse_ln   : per-query projection + bilinear gather + view fuse + post-LN
//  k4 mix_gemm  : y = h @ Wmix on v_wmma_f32_16x16x32_f16, writes transposed out
// ---------------------------------------------------------------------------

typedef __attribute__((ext_vector_type(16))) _Float16 v16h;
typedef __attribute__((ext_vector_type(8)))  _Float16 v8h;
typedef __attribute__((ext_vector_type(8)))  float    v8f;

#define BEV_H 50
#define BEV_W 50
#define NQ    (BEV_H * BEV_W)      // 2500
#define NQ_PAD 2512                 // next multiple of 16
#define NVIEW 6
#define NPATCH 37
#define NTOK  (NPATCH * NPATCH)     // 1369
#define CCH   768
#define CCTX  256
#define DPIL  4

// -------------------- block-wide (sum, sumsq) reduction ---------------------
__device__ inline void block_reduce2(float& a, float& b, float* sm) {
  const int lane = threadIdx.x & 31;
  const int wid  = threadIdx.x >> 5;
  for (int off = 16; off > 0; off >>= 1) {
    a += __shfl_down(a, off, 32);
    b += __shfl_down(b, off, 32);
  }
  if (lane == 0) { sm[wid * 2] = a; sm[wid * 2 + 1] = b; }
  __syncthreads();
  if (wid == 0) {
    const int nw = blockDim.x >> 5;
    a = (lane < nw) ? sm[lane * 2]     : 0.f;
    b = (lane < nw) ? sm[lane * 2 + 1] : 0.f;
    for (int off = 4; off > 0; off >>= 1) {
      a += __shfl_down(a, off, 32);
      b += __shfl_down(b, off, 32);
    }
    if (lane == 0) { sm[0] = a; sm[1] = b; }
  }
  __syncthreads();
  a = sm[0]; b = sm[1];
}

// -------------------- k1: densify grouped softmax mix -----------------------
// Wmix[c][cc] = softmax(logits[c/3])[c%3] if cc == c/3 else 0   (768 x 256)
__global__ __launch_bounds__(CCTX) void build_mix(const float* __restrict__ logits,
                                                  _Float16* __restrict__ wmix) {
  const int c  = blockIdx.x;    // 0..767
  const int cc = threadIdx.x;   // 0..255
  const int row = c / 3;
  const int j   = c - row * 3;
  float val = 0.f;
  if (cc == row) {
    const float l0 = logits[row * 3 + 0];
    const float l1 = logits[row * 3 + 1];
    const float l2 = logits[row * 3 + 2];
    const float m  = fmaxf(l0, fmaxf(l1, l2));
    const float e0 = __expf(l0 - m), e1 = __expf(l1 - m), e2 = __expf(l2 - m);
    const float ej = (j == 0) ? e0 : ((j == 1) ? e1 : e2);
    val = ej / (e0 + e1 + e2);
  }
  wmix[c * CCTX + cc] = (_Float16)val;
}

// -------------------- k2: token pre-LayerNorm (no affine), f32 -> f16 -------
__global__ __launch_bounds__(256) void ln_tokens(const float* __restrict__ tok,
                                                 _Float16* __restrict__ tokh) {
  __shared__ float sm[16];
  const int n = blockIdx.x;              // 0 .. 6*1369-1
  const int t = threadIdx.x;             // 0..255, 3 channels per thread
  const float* r = tok + (size_t)n * CCH;
  const float x0 = r[t], x1 = r[t + 256], x2 = r[t + 512];
  float s  = x0 + x1 + x2;
  float ss = x0 * x0 + x1 * x1 + x2 * x2;
  block_reduce2(s, ss, sm);
  const float mu   = s * (1.f / CCH);
  const float var  = ss * (1.f / CCH) - mu * mu;
  const float rstd = rsqrtf(var + 1e-5f);
  _Float16* w = tokh + (size_t)n * CCH;
  w[t]       = (_Float16)((x0 - mu) * rstd);
  w[t + 256] = (_Float16)((x1 - mu) * rstd);
  w[t + 512] = (_Float16)((x2 - mu) * rstd);
}

// -------------------- k3: projection + bilinear fuse + post-LN --------------
__global__ __launch_bounds__(256) void fuse_ln(const _Float16* __restrict__ tokh,
                                               const float* __restrict__ l2i,
                                               const float* __restrict__ w_view,
                                               const float* __restrict__ gamma,
                                               const float* __restrict__ beta,
                                               _Float16* __restrict__ hbuf) {
  __shared__ float sm[16];
  const int q = blockIdx.x;          // 0 .. 2511 (2500..2511 = zero padding)
  const int t = threadIdx.x;
  if (q >= NQ) {                     // uniform per block: safe early exit
    _Float16* w = hbuf + (size_t)q * CCH;
    w[t] = (_Float16)0.f; w[t + 256] = (_Float16)0.f; w[t + 512] = (_Float16)0.f;
    return;
  }
  const int qx = q % BEV_W, qy = q / BEV_W;
  const float X = (qx + 0.5f) * 2.048f - 51.2f;   // (x/50)*102.4 - 51.2
  const float Y = (qy + 0.5f) * 2.048f - 51.2f;

  float num0 = 0.f, num1 = 0.f, num2 = 0.f, den = 0.f;

  for (int v = 0; v < NVIEW; ++v) {
    const float* M = l2i + v * 16;
    const float m0 = M[0],  m1 = M[1],  m2 = M[2],  m3 = M[3];
    const float m4 = M[4],  m5 = M[5],  m6 = M[6],  m7 = M[7];
    const float m8 = M[8],  m9 = M[9],  mA = M[10], mB = M[11];
    const float wr = w_view[v];
    const float wv = (wr > 20.f) ? wr : log1pf(__expf(wr));   // softplus

    const _Float16* vbase = tokh + (size_t)v * NTOK * CCH;

    for (int d = 0; d < DPIL; ++d) {
      const float Z  = 0.5f + d * (7.f / 3.f) - 5.f;   // linspace(.5,7.5,4)-5
      const float px = fmaf(m0, X, fmaf(m1, Y, fmaf(m2, Z, m3)));
      const float py = fmaf(m4, X, fmaf(m5, Y, fmaf(m6, Z, m7)));
      const float pz = fmaf(m8, X, fmaf(m9, Y, fmaf(mA, Z, mB)));
      if (!(pz > 1e-5f)) continue;
      const float zi = 1.f / pz;
      const float u  = px * zi * (1.f / 518.f);
      const float vv = py * zi * (1.f / 518.f);
      if (!(u > 0.f && u < 1.f && vv > 0.f && vv < 1.f)) continue;
      den += wv;
      const float xp = (u  * 518.f + 0.5f) * (37.f / 518.f) - 0.5f;
      const float yp = (vv * 518.f + 0.5f) * (37.f / 518.f) - 0.5f;
      const float xf = floorf(xp), yf = floorf(yp);
      const float fx = xp - xf,    fy = yp - yf;
      const int ix = (int)xf, iy = (int)yf;
      #pragma unroll
      for (int dy = 0; dy < 2; ++dy) {
        #pragma unroll
        for (int dx = 0; dx < 2; ++dx) {
          const int xi = ix + dx, yi = iy + dy;
          if (xi < 0 || xi >= NPATCH || yi < 0 || yi >= NPATCH) continue;
          const float w = (dx ? fx : 1.f - fx) * (dy ? fy : 1.f - fy);
          if (w == 0.f) continue;
          const float ww = wv * w;
          const _Float16* row = vbase + (size_t)(yi * NPATCH + xi) * CCH + t;
          num0 = fmaf(ww, (float)row[0],   num0);
          num1 = fmaf(ww, (float)row[256], num1);
          num2 = fmaf(ww, (float)row[512], num2);
        }
      }
    }
  }

  const float inv = 1.f / (den + 1e-6f);
  const float f0 = num0 * inv, f1 = num1 * inv, f2 = num2 * inv;
  float s  = f0 + f1 + f2;
  float ss = f0 * f0 + f1 * f1 + f2 * f2;
  block_reduce2(s, ss, sm);
  const float mu   = s * (1.f / CCH);
  const float var  = ss * (1.f / CCH) - mu * mu;
  const float rstd = rsqrtf(var + 1e-5f);
  _Float16* w = hbuf + (size_t)q * CCH;
  w[t]       = (_Float16)((f0 - mu) * rstd * gamma[t]       + beta[t]);
  w[t + 256] = (_Float16)((f1 - mu) * rstd * gamma[t + 256] + beta[t + 256]);
  w[t + 512] = (_Float16)((f2 - mu) * rstd * gamma[t + 512] + beta[t + 512]);
}

// -------------------- k4: y = h @ Wmix via WMMA, transposed store -----------
// One wave32 per 16x16 output tile; K=768 in 24 steps of 32.
__global__ __launch_bounds__(32) void mix_gemm(const _Float16* __restrict__ h,
                                               const _Float16* __restrict__ wmix,
                                               float* __restrict__ out) {
  const int mq   = blockIdx.x * 16;        // query-tile base (0..2496)
  const int nb   = blockIdx.y * 16;        // cctx-tile base
  const int lane = threadIdx.x;            // 0..31
  const int half = lane >> 4;              // 0|1
  const int lr   = lane & 15;

  v8f acc = {};
  const _Float16* arow = h + (size_t)(mq + lr) * CCH;   // A: M = lane%16

  for (int kb = 0; kb < CCH; kb += 32) {
    // A 16x32 f16 fragment: lane half h holds K = {8h..8h+7, 16+8h..16+8h+7}
    const v8h alo = *(const v8h*)(arow + kb + 8 * half);
    const v8h ahi = *(const v8h*)(arow + kb + 16 + 8 * half);
    const v16h a = __builtin_shufflevector(alo, ahi,
        0, 1, 2, 3, 4, 5, 6, 7, 8, 9, 10, 11, 12, 13, 14, 15);
    // B 32x16 f16 fragment: lane holds column N = nb+lane%16, K = {8h..8h+7, 16+8h..16+8h+7}
    const _Float16* bp = wmix + (size_t)(kb + 8 * half) * CCTX + nb + lr;
    v16h b;
    #pragma unroll
    for (int s = 0; s < 16; ++s)
      b[s] = bp[(size_t)((s < 8) ? s : (s + 8)) * CCTX];
    acc = __builtin_amdgcn_wmma_f32_16x16x32_f16(
        /*neg_a=*/false, a, /*neg_b=*/false, b,
        /*c_mod=*/(short)0, acc, /*reuse_a=*/false, /*reuse_b=*/false);
  }

  // D layout: VGPR r -> M = r + 8*half, N = lane%16. Output is (256, 2500):
  // out[cc*2500 + q].
  const int cc = nb + lr;
  #pragma unroll
  for (int r = 0; r < 8; ++r) {
    const int m = mq + r + 8 * half;
    if (m < NQ) out[cc * NQ + m] = acc[r];
  }
}

// ---------------------------------------------------------------------------
extern "C" void kernel_launch(void* const* d_in, const int* in_sizes, int n_in,
                              void* d_out, int out_size, void* d_ws, size_t ws_size,
                              hipStream_t stream) {
  (void)in_sizes; (void)n_in; (void)out_size; (void)ws_size;
  const float* last_tokens = (const float*)d_in[0];  // (1,6,1369,768)
  const float* lidar2img   = (const float*)d_in[1];  // (1,6,4,4)
  const float* w_view      = (const float*)d_in[2];  // (1,6,1)
  const float* post_gamma  = (const float*)d_in[3];  // (768,)
  const float* post_beta   = (const float*)d_in[4];  // (768,)
  const float* logits      = (const float*)d_in[5];  // (256,3)
  float* out = (float*)d_out;                        // (1,256,50,50)

  // workspace layout (16B aligned slices)
  char* ws = (char*)d_ws;
  _Float16* tokh = (_Float16*)ws;                                   // 6*1369*768*2 = 12,616,704 B
  _Float16* wmix = (_Float16*)(ws + 12616704);                      //   768*256*2 =    393,216 B
  _Float16* hbuf = (_Float16*)(ws + 12616704 + 393216);             //  2512*768*2 =  3,858,432 B

  build_mix<<<dim3(CCH), dim3(CCTX), 0, stream>>>(logits, wmix);
  ln_tokens<<<dim3(NVIEW * NTOK), dim3(256), 0, stream>>>(last_tokens, tokh);
  fuse_ln<<<dim3(NQ_PAD), dim3(256), 0, stream>>>(tokh, lidar2img, w_view,
                                                  post_gamma, post_beta, hbuf);
  mix_gemm<<<dim3(NQ_PAD / 16, CCTX / 16), dim3(32), 0, stream>>>(hbuf, wmix, out);
}
